// YOSOFFN_69965017252065
// MI455X (gfx1250) — matrix operand order repaired and checked by reading, not compiled, not run on hardware
//
#include <hip/hip_runtime.h>
#include <hip/hip_bf16.h>

#define H 1024
#define I 4096
#define INV_PI 0.3183098861837906715f

typedef unsigned short u16;
typedef __attribute__((ext_vector_type(16))) __bf16 v16bf;
typedef __attribute__((ext_vector_type(8)))  float  v8f;

union FragBF {
    v16bf v;
    uint4 q[2];
};

__device__ __forceinline__ u16 f2bf(float x) {
    unsigned int u = __float_as_uint(x);
    u += 0x7FFFu + ((u >> 16) & 1u);   // round-to-nearest-even
    return (u16)(u >> 16);
}

// ---------------------------------------------------------------------------
// Prep 1: L2-normalize rows of k_weight [I,H] -> bf16 Kn [I,H]
// ---------------------------------------------------------------------------
__global__ void __launch_bounds__(256) knorm_kernel(const float* __restrict__ kw,
                                                    u16* __restrict__ Kn) {
    __shared__ float red[256];
    const int row = blockIdx.x, tid = threadIdx.x;
    const float* xr = kw + (size_t)row * H;
    float v0[4];
    float s2 = 0.f;
#pragma unroll
    for (int j = 0; j < 4; ++j) {
        float t = xr[tid + j * 256];
        v0[j] = t;
        s2 += t * t;
    }
    red[tid] = s2; __syncthreads();
    for (int st = 128; st > 0; st >>= 1) {
        if (tid < st) red[tid] += red[tid + st];
        __syncthreads();
    }
    const float scale = 1.0f / fmaxf(sqrtf(red[0]), 1e-12f);
    u16* o = Kn + (size_t)row * H;
#pragma unroll
    for (int j = 0; j < 4; ++j) o[tid + j * 256] = f2bf(v0[j] * scale);
}

// ---------------------------------------------------------------------------
// Prep 2: transpose+convert q_weight [I,H] f32 -> qT [H,I] bf16
// ---------------------------------------------------------------------------
__global__ void __launch_bounds__(256) qtrans_kernel(const float* __restrict__ qw,
                                                     u16* __restrict__ qT) {
    __shared__ float tile[32][33];
    const int i0 = blockIdx.x * 32, h0 = blockIdx.y * 32;
    const int tx = threadIdx.x, ty = threadIdx.y;
#pragma unroll
    for (int j = 0; j < 32; j += 8)
        tile[ty + j][tx] = qw[(size_t)(i0 + ty + j) * H + h0 + tx];
    __syncthreads();
#pragma unroll
    for (int j = 0; j < 32; j += 8)
        qT[(size_t)(h0 + ty + j) * I + i0 + tx] = f2bf(tile[tx][ty + j]);
}

// ---------------------------------------------------------------------------
// Prep 3: LayerNorm + L2 normalize rows of hidden [N,H] -> bf16 Qn [N,H]
// ---------------------------------------------------------------------------
__global__ void __launch_bounds__(256) ln_l2_kernel(const float* __restrict__ x,
                                                    const float* __restrict__ g,
                                                    const float* __restrict__ be,
                                                    u16* __restrict__ Qn) {
    __shared__ float red[256];
    const int row = blockIdx.x, tid = threadIdx.x;
    const float* xr = x + (size_t)row * H;
    float v0[4];
    float s = 0.f, s2 = 0.f;
#pragma unroll
    for (int j = 0; j < 4; ++j) {
        float t = xr[tid + j * 256];
        v0[j] = t;
        s += t;
        s2 += t * t;
    }
    red[tid] = s; __syncthreads();
    for (int st = 128; st > 0; st >>= 1) { if (tid < st) red[tid] += red[tid + st]; __syncthreads(); }
    const float sum = red[0]; __syncthreads();
    red[tid] = s2; __syncthreads();
    for (int st = 128; st > 0; st >>= 1) { if (tid < st) red[tid] += red[tid + st]; __syncthreads(); }
    const float sumsq = red[0]; __syncthreads();

    const float mu = sum * (1.0f / H);
    const float var = sumsq * (1.0f / H) - mu * mu;
    const float rstd = rsqrtf(var + 1e-12f);

    float y[4];
    float n2 = 0.f;
#pragma unroll
    for (int j = 0; j < 4; ++j) {
        const int c = tid + j * 256;
        float yy = (v0[j] - mu) * rstd * g[c] + be[c];
        y[j] = yy;
        n2 += yy * yy;
    }
    red[tid] = n2; __syncthreads();
    for (int st = 128; st > 0; st >>= 1) { if (tid < st) red[tid] += red[tid + st]; __syncthreads(); }
    const float scale = 1.0f / fmaxf(sqrtf(red[0]), 1e-12f);
    u16* o = Qn + (size_t)row * H;
#pragma unroll
    for (int j = 0; j < 4; ++j) o[tid + j * 256] = f2bf(y[j] * scale);
}

// ---------------------------------------------------------------------------
// Fused double-GEMM: per block of 16 rows:
//   loop over I in chunks of 128:
//     stage 1: S = Q(16x1024) x Kn_chunk^T   (one 16x16 tile per wave, bf16 WMMA)
//     map:     B = (1 - acos(clip(s))/pi)^9  -> bf16 into LDS
//     stage 2: X(16x1024) += B(16x128) x qT_chunk  (8 tiles per wave)
//   epilogue: row L2 norm + bias, f32 store.
// ---------------------------------------------------------------------------
__global__ void __launch_bounds__(256) yoso_fused_kernel(const u16* __restrict__ Qn,
                                                         const u16* __restrict__ Kn,
                                                         const u16* __restrict__ qT,
                                                         const float* __restrict__ bias,
                                                         float* __restrict__ out) {
    __shared__ __align__(16) u16 sQ[16 * 1032];   // Q tile, padded stride vs bank conflicts
    __shared__ __align__(16) u16 sB[16 * 136];    // B chunk [16 x 128], padded
    __shared__ float sRed[8 * 16];
    __shared__ float sNorm[16];

    const int tid   = threadIdx.x;
    const int w     = tid >> 5;        // wave id 0..7 (wave32)
    const int lane  = tid & 31;
    const int lrow  = lane & 15;
    const int lhalf = lane >> 4;
    const int n0    = blockIdx.x * 16;

    // preload 16 x 1024 bf16 Q tile into LDS
    {
        const u16* qg = Qn + (size_t)n0 * H;
        for (int idx = tid; idx < 16 * 128; idx += 256) {
            const int r = idx >> 7, c = (idx & 127) << 3;
            *(uint4*)(sQ + r * 1032 + c) = *(const uint4*)(qg + r * H + c);
        }
    }

    v8f xacc[8] = {};
    __syncthreads();

    for (int i0 = 0; i0 < I; i0 += 128) {
        // ---- stage 1: this wave's 16x16 S tile over K = 1024
        const u16* kb = Kn + (size_t)(i0 + w * 16 + lrow) * H;
        if (i0 + 128 < I)
            __builtin_prefetch((const void*)(kb + (size_t)128 * H), 0, 0);
        v8f sacc = {};
#pragma unroll 4
        for (int k = 0; k < H; k += 32) {
            FragBF a, b;
            const u16* qp = sQ + lrow * 1032 + k + lhalf * 8;   // A frag (16x32)
            a.q[0] = *(const uint4*)qp;
            a.q[1] = *(const uint4*)(qp + 16);
            const u16* kp = kb + k + lhalf * 8;                 // B frag (32x16)
            b.q[0] = *(const uint4*)kp;
            b.q[1] = *(const uint4*)(kp + 16);
            sacc = __builtin_amdgcn_wmma_f32_16x16x32_bf16(false, a.v, false, b.v,
                                                           (short)0, sacc, false, false);
        }
        __syncthreads();   // prior stage-2 readers are done with sB
        // ---- elementwise YOSO kernel map -> bf16 B chunk in LDS
#pragma unroll
        for (int v = 0; v < 8; ++v) {
            float s = sacc[v];
            s = fminf(fmaxf(s, -1.0f + 1e-6f), 1.0f - 1e-6f);
            const float bb = 1.0f - acosf(s) * INV_PI;
            const float b2 = bb * bb;
            const float b4 = b2 * b2;
            const float b8 = b4 * b4;
            sB[(lhalf * 8 + v) * 136 + w * 16 + lrow] = f2bf(b8 * bb);   // bb^9
        }
        __syncthreads();
        // ---- stage 2: X(16 x 128 cols of this wave) += B_chunk @ qT
#pragma unroll
        for (int ks = 0; ks < 128; ks += 32) {
            FragBF a;
            const u16* bp = sB + lrow * 136 + ks + lhalf * 8;   // A frag from LDS
            a.q[0] = *(const uint4*)bp;
            a.q[1] = *(const uint4*)(bp + 16);
#pragma unroll
            for (int t = 0; t < 8; ++t) {
                FragBF b;
                const u16* qp2 = qT + (size_t)(w * 128 + t * 16 + lrow) * I
                                    + (i0 + ks) + lhalf * 8;    // B frag (K-contig rows)
                b.q[0] = *(const uint4*)qp2;
                b.q[1] = *(const uint4*)(qp2 + 16);
                xacc[t] = __builtin_amdgcn_wmma_f32_16x16x32_bf16(false, a.v, false, b.v,
                                                                  (short)0, xacc[t], false, false);
            }
        }
    }

    // ---- epilogue: per-row L2 norm across all 1024 columns
    float ss[8];
#pragma unroll
    for (int v = 0; v < 8; ++v) {
        float acc = 0.f;
#pragma unroll
        for (int t = 0; t < 8; ++t) { const float x = xacc[t][v]; acc += x * x; }
        ss[v] = acc;
    }
#pragma unroll
    for (int d = 1; d < 16; d <<= 1) {
#pragma unroll
        for (int v = 0; v < 8; ++v) ss[v] += __shfl_xor(ss[v], d, 32);
    }
    if (lrow == 0) {
#pragma unroll
        for (int v = 0; v < 8; ++v) sRed[w * 16 + lhalf * 8 + v] = ss[v];
    }
    __syncthreads();
    if (tid < 16) {
        float acc = 0.f;
#pragma unroll
        for (int w2 = 0; w2 < 8; ++w2) acc += sRed[w2 * 16 + tid];
        sNorm[tid] = 1.0f / fmaxf(sqrtf(acc), 1e-12f);
    }
    __syncthreads();
#pragma unroll
    for (int t = 0; t < 8; ++t) {
        const int col = w * 128 + t * 16 + lrow;
        const float bv = bias[col];
#pragma unroll
        for (int v = 0; v < 8; ++v) {
            const int m = lhalf * 8 + v;
            out[(size_t)(n0 + m) * H + col] = xacc[t][v] * sNorm[m] + bv;
        }
    }
}

// ---------------------------------------------------------------------------
extern "C" void kernel_launch(void* const* d_in, const int* in_sizes, int n_in,
                              void* d_out, int out_size, void* d_ws, size_t ws_size,
                              hipStream_t stream) {
    const float* hidden = (const float*)d_in[0];
    const float* ln_w   = (const float*)d_in[1];
    const float* ln_b   = (const float*)d_in[2];
    const float* k_w    = (const float*)d_in[3];
    const float* q_w    = (const float*)d_in[4];
    const float* bias   = (const float*)d_in[5];
    float* out = (float*)d_out;

    const int N = in_sizes[0] / H;   // 8192

    // workspace layout: Qn [N,H] bf16 | Kn [I,H] bf16 | qT [H,I] bf16  (32 MB total)
    u16* Qn = (u16*)d_ws;
    u16* Kn = (u16*)((char*)d_ws + (size_t)N * H * sizeof(u16));
    u16* qT = (u16*)((char*)d_ws + (size_t)N * H * sizeof(u16) + (size_t)I * H * sizeof(u16));

    knorm_kernel<<<I, 256, 0, stream>>>(k_w, Kn);
    qtrans_kernel<<<dim3(I / 32, H / 32), dim3(32, 8), 0, stream>>>(q_w, qT);
    ln_l2_kernel<<<N, 256, 0, stream>>>(hidden, ln_w, ln_b, Qn);
    yoso_fused_kernel<<<N / 16, 256, 0, stream>>>(Qn, Kn, qT, bias, out);
}
